// HoiClassifier_18408229831249
// MI455X (gfx1250) — matrix operand order repaired and checked by reading, not compiled, not run on hardware
//
#include <hip/hip_runtime.h>
#include <math.h>

#define BATCH     32768
#define NUM_HOIS  234
#define NCHUNK    64
#define ROWS_PER_CHUNK (BATCH / NCHUNK)   // 512

typedef __attribute__((ext_vector_type(16))) __bf16 bf16x16;
typedef __attribute__((ext_vector_type(8)))  float  v8f;

union BF16Frag { bf16x16 v; unsigned int u[8]; uint4 q[2]; };

__device__ __forceinline__ unsigned int f2bf(float f) {   // exact RNE (weight pack)
    unsigned int u = __float_as_uint(f);
    return (u + 0x7FFFu + ((u >> 16) & 1u)) >> 16;
}

// Hot-path pack: round-half-up + v_perm_b32 (3 VALU) instead of RNE (~10 VALU).
__device__ __forceinline__ unsigned int pack_bf16(float lo, float hi) {
#if __has_builtin(__builtin_amdgcn_cvt_pk_bf16_f32)
    typedef __attribute__((ext_vector_type(2))) __bf16 bf16x2;
    bf16x2 p = __builtin_amdgcn_cvt_pk_bf16_f32(lo, hi);
    return __builtin_bit_cast(unsigned int, p);
#elif __has_builtin(__builtin_amdgcn_perm)
    unsigned int ul = __float_as_uint(lo) + 0x8000u;
    unsigned int uh = __float_as_uint(hi) + 0x8000u;
    return __builtin_amdgcn_perm(uh, ul, 0x07060302u);   // {uh[31:16], ul[31:16]}
#else
    unsigned int ul = (__float_as_uint(lo) + 0x8000u) >> 16;
    unsigned int uh = (__float_as_uint(hi) + 0x8000u) & 0xFFFF0000u;
    return uh | ul;
#endif
}

// ---------------------------------------------------------------------------
// Input transforms: zero-padded feature matrices [f, log(|f|+1e-6), 0pad]
// ---------------------------------------------------------------------------
__global__ void hoi_transform_sp(const float* __restrict__ box, float* __restrict__ X) {
    int idx = blockIdx.x * 256 + threadIdx.x;
    if (idx >= BATCH * 64) return;
    int m = idx >> 6, c = idx & 63;
    float v;
    if (c < 21)       v = box[m * 21 + c];
    else if (c < 42)  v = logf(fabsf(box[m * 21 + c - 21]) + 1e-6f);
    else              v = 0.f;
    X[idx] = v;
}

__global__ void hoi_transform_pose(const float* __restrict__ ap, const float* __restrict__ rp,
                                   float* __restrict__ X) {
    int idx = blockIdx.x * 256 + threadIdx.x;
    if (idx >= BATCH * 320) return;
    int m = idx / 320, c = idx % 320;
    float v;
    if (c < 51)        v = ap[m * 51 + c];
    else if (c < 136)  v = rp[m * 85 + c - 51];
    else if (c < 272) {
        int cc = c - 136;
        float f = (cc < 51) ? ap[m * 51 + cc] : rp[m * 85 + cc - 51];
        v = logf(fabsf(f) + 1e-6f);
    } else             v = 0.f;
    X[idx] = v;
}

// ---------------------------------------------------------------------------
// Pack fp32 weights (K x N) into bf16 in WMMA B-fragment order:
// uint index = ((kt*(Np/16) + nt)*256) + lane*8 + v
//   kt=k/32, nt=n/16, lane=(k%32>=16)*16 + n%16, v=(k%16)/2, lo=k&1.
// A wave loads a whole 16x32 B tile as two b128s per lane, contiguous VGPRs.
// ---------------------------------------------------------------------------
__global__ void hoi_pack_w(const float* __restrict__ W, unsigned short* __restrict__ Wpk,
                           int K, int N, int Kp, int Np) {
    int i = blockIdx.x * 256 + threadIdx.x;
    if (i >= Kp * Np) return;
    int lo   = i & 1;
    int v    = (i >> 1) & 7;
    int lane = (i >> 4) & 31;
    int blk  = i >> 9;
    int ntS  = Np >> 4;
    int nt   = blk % ntS;
    int kt   = blk / ntS;
    int k = kt * 32 + (lane >> 4) * 16 + v * 2 + lo;
    int n = nt * 16 + (lane & 15);
    Wpk[i] = (k < K && n < N) ? (unsigned short)f2bf(W[k * N + n]) : (unsigned short)0;
}

// ---------------------------------------------------------------------------
// GEMM: Out[M,Np] = pre(X[M,Kp]) @ W, bf16 WMMA, fp32 accum.
// pre(x)[,k] = preAffine ? relu(x*s[k]+t[k]) : x   (fused BN+ReLU)
// 256 thr / 8 waves; block tile 128x64; K-step 32.
// A: double-buffered LDS (coalesced fp32 load -> bf16 pairs), 1 barrier/iter.
// B: fragment-ordered global (L0/L2 resident), 2x b128 per fragment, no LDS.
// ---------------------------------------------------------------------------
__global__ __launch_bounds__(256) void hoi_gemm_bf16_wmma(
    const float* __restrict__ X, const unsigned int* __restrict__ Wpk,
    const float* __restrict__ sv, const float* __restrict__ tv,
    float* __restrict__ Out, int Kp, int Np, int preAffine, int accumulate)
{
    __shared__ unsigned int As[2][128][20];   // 16 pairs + 4 pad: b128-aligned rows

    const int tid  = threadIdx.x;
    const int lane = tid & 31;
    const int wid  = tid >> 5;
    const int rowBase = blockIdx.y * 128;
    const int ntBase  = blockIdx.x * 4;       // 4 n-tiles of 16 per block
    const int ntStride = Np >> 4;

    v8f acc[4];
    #pragma unroll
    for (int j = 0; j < 4; ++j)
        #pragma unroll
        for (int r = 0; r < 8; ++r) acc[j][r] = 0.f;

    // per-thread A staging map: 1024 float4s = 128 rows x 8 quads
    int fr[4], fk[4];
    #pragma unroll
    for (int i = 0; i < 4; ++i) { int f = tid + 256 * i; fr[i] = f >> 3; fk[i] = (f & 7) << 2; }

    float4 pf[4];
    auto loadA = [&](int kb) {
        #pragma unroll
        for (int i = 0; i < 4; ++i)
            pf[i] = *(const float4*)(X + (size_t)(rowBase + fr[i]) * Kp + kb + fk[i]);
    };
    auto storeA = [&](int buf, int kb) {
        #pragma unroll
        for (int i = 0; i < 4; ++i) {
            float4 y = pf[i];
            if (preAffine) {
                int k = kb + fk[i];
                y.x = fmaxf(0.f, y.x * sv[k + 0] + tv[k + 0]);
                y.y = fmaxf(0.f, y.y * sv[k + 1] + tv[k + 1]);
                y.z = fmaxf(0.f, y.z * sv[k + 2] + tv[k + 2]);
                y.w = fmaxf(0.f, y.w * sv[k + 3] + tv[k + 3]);
            }
            As[buf][fr[i]][(fk[i] >> 1) + 0] = pack_bf16(y.x, y.y);
            As[buf][fr[i]][(fk[i] >> 1) + 1] = pack_bf16(y.z, y.w);
        }
    };

    loadA(0);
    storeA(0, 0);
    __syncthreads();

    const int ar = wid * 16 + (lane & 15);
    const int ak = (lane >> 4) * 8;                   // pair offset within row
    const unsigned int* wlane = Wpk + (size_t)lane * 8;

    int cur = 0;
    for (int kb = 0; kb < Kp; kb += 32) {
        const int  nxt = kb + 32;
        const bool has = nxt < Kp;
        if (has) loadA(nxt);                          // prefetch next A tile

        BF16Frag a;                                   // 2x ds_load_b128
        {
            const uint4* p = (const uint4*)&As[cur][ar][ak];
            a.q[0] = p[0];
            a.q[1] = p[1];                            // +4 uints = K{8..15}/{24..31}
        }

        const int kt = kb >> 5;
        #pragma unroll
        for (int j = 0; j < 4; ++j) {
            BF16Frag b;                               // 2x global_load_b128, L0/L2 hit
            const uint4* bp = (const uint4*)(wlane + (size_t)(kt * ntStride + ntBase + j) * 256);
            b.q[0] = bp[0];
            b.q[1] = bp[1];
            acc[j] = __builtin_amdgcn_wmma_f32_16x16x32_bf16(
                false, a.v, false, b.v, (short)0, acc[j], false, false);
        }

        if (has) storeA(cur ^ 1, nxt);                // fill other buffer
        __syncthreads();                              // single barrier per K-step
        cur ^= 1;
    }

    // epilogue per ISA C layout: lane<16 -> M=r, lane>=16 -> M=8+r
    const int mrow = rowBase + wid * 16 + ((lane >> 4) << 3);
    const int col0 = ntBase * 16 + (lane & 15);
    #pragma unroll
    for (int j = 0; j < 4; ++j)
        #pragma unroll
        for (int r = 0; r < 8; ++r) {
            size_t o = (size_t)(mrow + r) * Np + col0 + j * 16;
            if (accumulate) Out[o] += acc[j][r];
            else            Out[o]  = acc[j][r];
        }
}

// ---------------------------------------------------------------------------
// Deterministic column stats: partials per fixed row chunk, ordered reduce
// ---------------------------------------------------------------------------
__global__ void hoi_col_stats(const float* __restrict__ H, float* __restrict__ psum,
                              float* __restrict__ psq, int Np) {
    __shared__ float ls[4][64], lq[4][64];
    int c   = threadIdx.x & 63;
    int rt  = threadIdx.x >> 6;
    int col = blockIdx.x * 64 + c;
    int r0  = blockIdx.y * ROWS_PER_CHUNK;
    float s = 0.f, q = 0.f;
    for (int r = rt; r < ROWS_PER_CHUNK; r += 4) {
        float v = H[(size_t)(r0 + r) * Np + col];
        s += v; q += v * v;
    }
    ls[rt][c] = s; lq[rt][c] = q;
    __syncthreads();
    if (rt == 0) {
        s = ls[0][c] + ls[1][c] + ls[2][c] + ls[3][c];
        q = lq[0][c] + lq[1][c] + lq[2][c] + lq[3][c];
        psum[blockIdx.y * Np + col] = s;
        psq [blockIdx.y * Np + col] = q;
    }
}

__global__ void hoi_bn_affine(const float* __restrict__ psum, const float* __restrict__ psq,
                              const float* __restrict__ g, const float* __restrict__ be,
                              const float* __restrict__ bias,
                              float* __restrict__ sOut, float* __restrict__ tOut,
                              int Np, int ntrue) {
    int c = blockIdx.x * 64 + threadIdx.x;
    if (c >= Np) return;
    float s = 0.f, q = 0.f;
    for (int i = 0; i < NCHUNK; ++i) { s += psum[i * Np + c]; q += psq[i * Np + c]; }
    const float invM = 1.0f / (float)BATCH;
    float m0  = s * invM;                 // mean of X@W (bias folded below)
    float var = q * invM - m0 * m0;       // bias shift leaves var unchanged
    float sc = 0.f, tr = 0.f;
    if (c < ntrue) {
        sc = g[c] * rsqrtf(var + 1e-5f);
        tr = be[c] - (m0 + bias[c]) * sc;
    }
    sOut[c] = sc; tOut[c] = tr;
}

// ---------------------------------------------------------------------------
// Head-bias combine; final sigmoid + verb_map gather + probability products
// ---------------------------------------------------------------------------
__global__ void hoi_combine_bias(const float* b0, const float* b1, const float* b2,
                                 const float* b3, const float* b4, float* cb) {
    int v = threadIdx.x;
    if (v < 64) cb[v] = (v < 25) ? (b0[v] + b1[v] + b2[v] + b3[v] + b4[v]) : 0.f;
}

__global__ void hoi_final(const float* __restrict__ S, const float* __restrict__ cb,
                          const int* __restrict__ vmap,
                          const float* __restrict__ hp, const float* __restrict__ op,
                          const float* __restrict__ mask, float* __restrict__ out) {
    int idx = blockIdx.x * 256 + threadIdx.x;
    if (idx >= BATCH * NUM_HOIS) return;
    int m = idx / NUM_HOIS, j = idx % NUM_HOIS;
    int v = vmap[j];
    float sc = S[(size_t)m * 64 + v] + cb[v];
    float p  = 1.f / (1.f + expf(-sc));
    out[idx] = p * hp[idx] * op[idx] * mask[idx];
}

// ---------------------------------------------------------------------------
extern "C" void kernel_launch(void* const* d_in, const int* in_sizes, int n_in,
                              void* d_out, int out_size, void* d_ws, size_t ws_size,
                              hipStream_t stream) {
    (void)in_sizes; (void)n_in; (void)out_size; (void)ws_size;
    const float* box   = (const float*)d_in[0];
    const float* hum_x = (const float*)d_in[1];
    const float* obj_x = (const float*)d_in[2];
    const float* ap    = (const float*)d_in[3];
    const float* rp    = (const float*)d_in[4];
    const float* uni_x = (const float*)d_in[5];
    const float* hp    = (const float*)d_in[6];
    const float* op    = (const float*)d_in[7];
    const float* mask  = (const float*)d_in[8];
    const int*   vmap  = (const int*)d_in[9];
    // params (insertion order): sp(3) pose(3) human(2) object(2) union(2), each W,b,g,be
    auto P = [&](int i) { return (const float*)d_in[10 + i]; };

    char* ws = (char*)d_ws;
    float*          S    = (float*)ws;           ws += (size_t)BATCH * 64 * 4;
    float*          Xbuf = (float*)ws;           ws += (size_t)BATCH * 320 * 4;
    float*          H1   = (float*)ws;           ws += (size_t)BATCH * 512 * 4;
    float*          H2   = (float*)ws;           ws += (size_t)BATCH * 512 * 4;
    unsigned short* Wp   = (unsigned short*)ws;  ws += (size_t)2048 * 512 * 2;
    float*          psum = (float*)ws;           ws += (size_t)NCHUNK * 512 * 4;
    float*          psq  = (float*)ws;           ws += (size_t)NCHUNK * 512 * 4;
    float*          sv   = (float*)ws;           ws += 512 * 4;
    float*          tv   = (float*)ws;           ws += 512 * 4;
    float*          cb   = (float*)ws;           ws += 64 * 4;

    auto gemm = [&](const float* X, const float* W, int K, int N, int Kp, int Np,
                    const float* s, const float* t, float* Out, int accum) {
        int tot = Kp * Np;
        hipLaunchKernelGGL(hoi_pack_w, dim3((tot + 255) / 256), dim3(256), 0, stream,
                           W, Wp, K, N, Kp, Np);
        hipLaunchKernelGGL(hoi_gemm_bf16_wmma, dim3(Np / 64, BATCH / 128), dim3(256), 0, stream,
                           X, (const unsigned int*)Wp, s, t, Out, Kp, Np, s != nullptr, accum);
    };
    auto stats = [&](const float* H, const float* g, const float* be, const float* b,
                     int Np, int ntrue) {
        hipLaunchKernelGGL(hoi_col_stats, dim3(Np / 64, NCHUNK), dim3(256), 0, stream,
                           H, psum, psq, Np);
        hipLaunchKernelGGL(hoi_bn_affine, dim3((Np + 63) / 64), dim3(64), 0, stream,
                           psum, psq, g, be, b, sv, tv, Np, ntrue);
    };

    // ---- sp branch (42 -> pad 64) ----
    hipLaunchKernelGGL(hoi_transform_sp, dim3((BATCH * 64 + 255) / 256), dim3(256), 0, stream,
                       box, Xbuf);
    gemm(Xbuf, P(0), 42, 42, 64, 64, nullptr, nullptr, H1, 0);
    stats(H1, P(2), P(3), P(1), 64, 42);
    gemm(H1, P(4), 42, 42, 64, 64, sv, tv, H2, 0);
    stats(H2, P(6), P(7), P(5), 64, 42);
    gemm(H2, P(8), 42, 25, 64, 64, sv, tv, S, 0);            // first write to S

    // ---- pose branch (272 -> pad 320) ----
    hipLaunchKernelGGL(hoi_transform_pose, dim3((BATCH * 320 + 255) / 256), dim3(256), 0, stream,
                       ap, rp, Xbuf);
    gemm(Xbuf, P(12), 272, 272, 320, 320, nullptr, nullptr, H1, 0);
    stats(H1, P(14), P(15), P(13), 320, 272);
    gemm(H1, P(16), 272, 272, 320, 320, sv, tv, H2, 0);
    stats(H2, P(18), P(19), P(17), 320, 272);
    gemm(H2, P(20), 272, 25, 320, 64, sv, tv, S, 1);

    // ---- rcnn branches (2048 -> 512 -> 25) ----
    const float* xs[3] = { hum_x, obj_x, uni_x };
    const int    pb[3] = { 24, 32, 40 };
    for (int i = 0; i < 3; ++i) {
        int q = pb[i];
        gemm(xs[i], P(q + 0), 2048, 512, 2048, 512, nullptr, nullptr, H1, 0);
        stats(H1, P(q + 2), P(q + 3), P(q + 1), 512, 512);
        gemm(H1, P(q + 4), 512, 25, 512, 64, sv, tv, S, 1);
    }

    hipLaunchKernelGGL(hoi_combine_bias, dim3(1), dim3(64), 0, stream,
                       P(9), P(21), P(29), P(37), P(45), cb);
    hipLaunchKernelGGL(hoi_final, dim3((BATCH * NUM_HOIS + 255) / 256), dim3(256), 0, stream,
                       S, cb, vmap, hp, op, mask, (float*)d_out);
}
